// MedPredictor_34883724378627
// MI455X (gfx1250) — matrix operand order repaired and checked by reading, not compiled, not run on hardware
//
#include <hip/hip_runtime.h>
#include <cstdint>
#include <cmath>

// ---------------------------------------------------------------------------
// Types for CDNA5 WMMA (wave32): v_wmma_f32_16x16x32_bf16
// ---------------------------------------------------------------------------
typedef __bf16 bf16;
typedef __attribute__((ext_vector_type(16))) __bf16 v16bf;
typedef __attribute__((ext_vector_type(8)))  __bf16 v8bf;
typedef __attribute__((ext_vector_type(8)))  float  v8f;

constexpr int kN   = 32768;   // nodes
constexpr int kE   = 131072;  // directed edges (before self loops)
constexpr int kB   = 32;      // graphs
constexpr int kEMB = 512;
constexpr int kOUT = 2048;    // EMB * HEADS
constexpr int kH   = 4;

// ---------------------------------------------------------------------------
// float atomic max via sign-aware integer atomics (works with -inf init)
// ---------------------------------------------------------------------------
__device__ __forceinline__ void atomicMaxF(float* addr, float v) {
  if (v >= 0.0f) atomicMax((int*)addr, __float_as_int(v));
  else           atomicMin((unsigned int*)addr, __float_as_uint(v));
}

// ---------------------------------------------------------------------------
// K0: transpose + convert weights to bf16:  Wt[n][k] = (bf16) W[k][n]
// ---------------------------------------------------------------------------
__global__ void transpose_cvt(const float* __restrict__ W, bf16* __restrict__ Wt,
                              int K, int Nn) {
  int idx = blockIdx.x * blockDim.x + threadIdx.x;
  if (idx >= K * Nn) return;
  int n = idx / K;
  int k = idx - n * K;
  Wt[(size_t)n * K + k] = (bf16)W[(size_t)k * Nn + n];
}

// ---------------------------------------------------------------------------
// K1: embedding gather + fp32 -> bf16 convert.  One thread = 8 channels.
// ---------------------------------------------------------------------------
__global__ __launch_bounds__(256) void gather_embed(const int* __restrict__ code,
                                                    const float* __restrict__ emb,
                                                    bf16* __restrict__ x0) {
  size_t tid = (size_t)blockIdx.x * blockDim.x + threadIdx.x;  // kN*512/8
  int n  = (int)(tid >> 6);           // 64 chunks of 8 per node
  int c8 = (int)(tid & 63) << 3;
  const float* src = emb + (size_t)code[n] * kEMB + c8;
  float4 f0 = *(const float4*)(src);
  float4 f1 = *(const float4*)(src + 4);
  v8bf o;
  o[0] = (bf16)f0.x; o[1] = (bf16)f0.y; o[2] = (bf16)f0.z; o[3] = (bf16)f0.w;
  o[4] = (bf16)f1.x; o[5] = (bf16)f1.y; o[6] = (bf16)f1.z; o[7] = (bf16)f1.w;
  *(v8bf*)(x0 + (size_t)n * kEMB + c8) = o;
}

// ---------------------------------------------------------------------------
// Fragment loader: two 16B halves -> one 16-element bf16 fragment
// ---------------------------------------------------------------------------
__device__ __forceinline__ v16bf load_frag_pair(const bf16* __restrict__ p0,
                                                const bf16* __restrict__ p1) {
  v8bf lo = *(const v8bf*)p0;
  v8bf hi = *(const v8bf*)p1;
  v16bf f;
#pragma unroll
  for (int i = 0; i < 8; ++i) { f[i] = lo[i]; f[i + 8] = hi[i]; }
  return f;
}

// ---------------------------------------------------------------------------
// K2/K3: WMMA bf16 GEMM  C[M,Nn] = A[M,K] * Bt[Nn,K]^T (+ bias), bf16 out.
// Block = 8 waves; each wave computes a 16x64 strip (A frag reused x4).
// Software-pipelined: fragments for k+32 are loaded before the WMMAs for k,
// so VMEM overlaps the matrix pipe instead of s_wait_loadcnt 0 per WMMA.
// ---------------------------------------------------------------------------
__global__ __launch_bounds__(256) void wmma_gemm_bf16(
    const bf16* __restrict__ A,     // [M,K] row-major, bf16
    const bf16* __restrict__ Bt,    // [Nn,K] row-major (= B transposed), bf16
    const float* __restrict__ bias, // [Nn] or nullptr
    bf16* __restrict__ C,           // [M,Nn] row-major, bf16
    int M, int Nn, int K) {
  const int lane = threadIdx.x & 31;
  const int wave = threadIdx.x >> 5;
  const int mi = wave >> 2;                 // 0..1  (M sub-tile)
  const int ni = wave & 3;                  // 0..3  (N sub-tile)
  const int rowBase = blockIdx.x * 32 + mi * 16;
  const int colBase = blockIdx.y * 256 + ni * 64;

  const int r  = lane & 15;
  const int hi = lane >> 4;                 // 0 or 1 (half-wave)

  // A fragment: lanes 0-15 -> K = {k0..k0+15}
  //             lanes 16-31 -> K = {k0+16..k0+31}
  const bf16* aBase = A + (size_t)(rowBase + r) * K + hi * 16;
  // B fragment: col = colBase + t*16 + r, 16 contiguous K at hi*16
  const bf16* bBase[4];
#pragma unroll
  for (int t = 0; t < 4; ++t)
    bBase[t] = Bt + (size_t)(colBase + t * 16 + r) * K + hi * 16;

  v8f acc[4] = {};

  // prologue: fragments for k0 = 0
  v16bf aCur = load_frag_pair(aBase, aBase + 8);
  v16bf bCur0 = load_frag_pair(bBase[0], bBase[0] + 8);
  v16bf bCur1 = load_frag_pair(bBase[1], bBase[1] + 8);
  v16bf bCur2 = load_frag_pair(bBase[2], bBase[2] + 8);
  v16bf bCur3 = load_frag_pair(bBase[3], bBase[3] + 8);

  for (int k0 = 0; k0 < K - 32; k0 += 32) {
    // issue next iteration's loads first (double buffer)
    const int kn = k0 + 32;
    v16bf aNxt  = load_frag_pair(aBase + kn, aBase + kn + 8);
    v16bf bNxt0 = load_frag_pair(bBase[0] + kn, bBase[0] + kn + 8);
    v16bf bNxt1 = load_frag_pair(bBase[1] + kn, bBase[1] + kn + 8);
    v16bf bNxt2 = load_frag_pair(bBase[2] + kn, bBase[2] + kn + 8);
    v16bf bNxt3 = load_frag_pair(bBase[3] + kn, bBase[3] + kn + 8);

    acc[0] = __builtin_amdgcn_wmma_f32_16x16x32_bf16(false, aCur, false, bCur0,
                                                     (short)0, acc[0], false, false);
    acc[1] = __builtin_amdgcn_wmma_f32_16x16x32_bf16(false, aCur, false, bCur1,
                                                     (short)0, acc[1], false, false);
    acc[2] = __builtin_amdgcn_wmma_f32_16x16x32_bf16(false, aCur, false, bCur2,
                                                     (short)0, acc[2], false, false);
    acc[3] = __builtin_amdgcn_wmma_f32_16x16x32_bf16(false, aCur, false, bCur3,
                                                     (short)0, acc[3], false, false);

    aCur = aNxt;
    bCur0 = bNxt0; bCur1 = bNxt1; bCur2 = bNxt2; bCur3 = bNxt3;
  }

  // epilogue: last k-step
  acc[0] = __builtin_amdgcn_wmma_f32_16x16x32_bf16(false, aCur, false, bCur0,
                                                   (short)0, acc[0], false, false);
  acc[1] = __builtin_amdgcn_wmma_f32_16x16x32_bf16(false, aCur, false, bCur1,
                                                   (short)0, acc[1], false, false);
  acc[2] = __builtin_amdgcn_wmma_f32_16x16x32_bf16(false, aCur, false, bCur2,
                                                   (short)0, acc[2], false, false);
  acc[3] = __builtin_amdgcn_wmma_f32_16x16x32_bf16(false, aCur, false, bCur3,
                                                   (short)0, acc[3], false, false);

  // C/D f32 layout: element v -> row = rowBase + v + hi*8, col = colBase+t*16+r
#pragma unroll
  for (int t = 0; t < 4; ++t) {
    const int col = colBase + t * 16 + r;
    const float bv = bias ? bias[col] : 0.0f;
#pragma unroll
    for (int v = 0; v < 8; ++v) {
      const int row = rowBase + v + hi * 8;
      C[(size_t)row * Nn + col] = (bf16)(acc[t][v] + bv);
    }
  }
}

// ---------------------------------------------------------------------------
// K4: attention coefficients: a_src[n,h] = <h[n, h*512:], att_src[h]>, ditto dst
// One wave per (node, head).
// ---------------------------------------------------------------------------
__global__ __launch_bounds__(256) void attn_coef(const bf16* __restrict__ h,
                                                 const float* __restrict__ att_src,
                                                 const float* __restrict__ att_dst,
                                                 float* __restrict__ a_src,
                                                 float* __restrict__ a_dst) {
  int gw = blockIdx.x * 8 + (threadIdx.x >> 5);   // over kN*kH
  int lane = threadIdx.x & 31;
  int n = gw >> 2, hd = gw & 3;
  const bf16*  hp = h + (size_t)n * kOUT + hd * kEMB;
  const float* as = att_src + hd * kEMB;
  const float* ad = att_dst + hd * kEMB;
  float s = 0.f, d = 0.f;
  for (int c = lane; c < kEMB; c += 32) {
    float hv = (float)hp[c];
    s += hv * as[c];
    d += hv * ad[c];
  }
#pragma unroll
  for (int off = 16; off; off >>= 1) {
    s += __shfl_xor(s, off, 32);
    d += __shfl_xor(d, off, 32);
  }
  if (lane == 0) { a_src[gw] = s; a_dst[gw] = d; }
}

// ---------------------------------------------------------------------------
// K5: initialize accumulators (out=0, amax=-inf, denom=0)
// ---------------------------------------------------------------------------
__global__ void init_ws(float* __restrict__ out, float* __restrict__ amax,
                        float* __restrict__ denom) {
  size_t i = (size_t)blockIdx.x * blockDim.x + threadIdx.x;
  if (i < (size_t)kN * kOUT) out[i] = 0.0f;
  if (i < (size_t)kN * kH) { amax[i] = -INFINITY; denom[i] = 0.0f; }
}

// ---------------------------------------------------------------------------
// K6/K7: segment max and exp-sum over dst (thread per (edge, head))
// edges e < kE from edge_index; e >= kE are self loops.
// ---------------------------------------------------------------------------
__device__ __forceinline__ void edge_sd(const int* ei, int e, int& s, int& d) {
  if (e < kE) { s = ei[e]; d = ei[kE + e]; } else { s = d = e - kE; }
}

__global__ void edge_max(const int* __restrict__ ei, const float* __restrict__ a_src,
                         const float* __restrict__ a_dst, float* __restrict__ amax) {
  int t = blockIdx.x * blockDim.x + threadIdx.x;
  if (t >= (kE + kN) * kH) return;
  int e = t >> 2, hd = t & 3, s, d;
  edge_sd(ei, e, s, d);
  float l = a_src[s * kH + hd] + a_dst[d * kH + hd];
  l = l > 0.f ? l : 0.2f * l;
  atomicMaxF(&amax[d * kH + hd], l);
}

__global__ void edge_expsum(const int* __restrict__ ei, const float* __restrict__ a_src,
                            const float* __restrict__ a_dst, const float* __restrict__ amax,
                            float* __restrict__ denom) {
  int t = blockIdx.x * blockDim.x + threadIdx.x;
  if (t >= (kE + kN) * kH) return;
  int e = t >> 2, hd = t & 3, s, d;
  edge_sd(ei, e, s, d);
  float l = a_src[s * kH + hd] + a_dst[d * kH + hd];
  l = l > 0.f ? l : 0.2f * l;
  atomicAdd(&denom[d * kH + hd], __expf(l - amax[d * kH + hd]));
}

// ---------------------------------------------------------------------------
// K8: message aggregation. One wave per (edge, head); h gather hits L2 (bf16).
// ---------------------------------------------------------------------------
__global__ __launch_bounds__(256) void edge_aggregate(
    const int* __restrict__ ei, const float* __restrict__ a_src,
    const float* __restrict__ a_dst, const float* __restrict__ amax,
    const float* __restrict__ denom, const bf16* __restrict__ h,
    float* __restrict__ out) {
  int gw = blockIdx.x * 8 + (threadIdx.x >> 5);   // over (kE+kN)*kH
  int lane = threadIdx.x & 31;
  int e = gw >> 2, hd = gw & 3, s, d;
  edge_sd(ei, e, s, d);
  float l = a_src[s * kH + hd] + a_dst[d * kH + hd];
  l = l > 0.f ? l : 0.2f * l;
  float alpha = __expf(l - amax[d * kH + hd]) / (denom[d * kH + hd] + 1e-16f);
  const bf16* hp = h + (size_t)s * kOUT + hd * kEMB;
  float* op = out + (size_t)d * kOUT + hd * kEMB;
  for (int c = lane; c < kEMB; c += 32)
    atomicAdd(&op[c], alpha * (float)hp[c]);
}

// ---------------------------------------------------------------------------
// K9: atomic-free pooling.  batch is SORTED, so each graph's nodes are a
// contiguous range: binary-search [lo,hi) per graph, reduce serially per
// channel with fully coalesced reads.  Folds +bias_gat, leaky_relu(0.01),
// and the mean division.  grid = (graph, channel_chunk).
// ---------------------------------------------------------------------------
__device__ __forceinline__ int lower_bound_i(const int* __restrict__ a, int n, int key) {
  int lo = 0, hi = n;
  while (lo < hi) { int mid = (lo + hi) >> 1; if (a[mid] < key) lo = mid + 1; else hi = mid; }
  return lo;
}

__global__ __launch_bounds__(256) void pool_by_graph(
    const float* __restrict__ out, const float* __restrict__ bias_gat,
    const int* __restrict__ batch, float* __restrict__ pmean,
    float* __restrict__ pmax) {
  int b = blockIdx.x;
  int c = blockIdx.y * 256 + threadIdx.x;
  int lo  = lower_bound_i(batch, kN, b);
  int hi2 = lower_bound_i(batch, kN, b + 1);
  float bv = bias_gat[c];
  float s = 0.f, m = -INFINITY;
  for (int n = lo; n < hi2; ++n) {
    float v = out[(size_t)n * kOUT + c] + bv;
    v = v > 0.f ? v : 0.01f * v;
    s += v;
    m = m > v ? m : v;
  }
  float cnt = (float)(hi2 - lo);
  pmean[(size_t)b * kOUT + c] = s / (cnt > 1.f ? cnt : 1.f);
  pmax [(size_t)b * kOUT + c] = m;
}

// ---------------------------------------------------------------------------
// K10: MLP heads. One wave per (graph, task); lane j owns hidden unit j (HID=32).
// ---------------------------------------------------------------------------
__global__ __launch_bounds__(256) void mlp_heads(
    const float* __restrict__ pmean, const float* __restrict__ pmax,
    const float* __restrict__ W1r, const float* __restrict__ b1r,
    const float* __restrict__ W2r, const float* __restrict__ b2r,
    const float* __restrict__ W1m, const float* __restrict__ b1m,
    const float* __restrict__ W2m, const float* __restrict__ b2m,
    float* __restrict__ outp) {
  int gw = blockIdx.x * 8 + (threadIdx.x >> 5);   // 0..63
  int lane = threadIdx.x & 31;
  int task = gw >> 5;           // 0 = readmit, 1 = mortal
  int b = gw & 31;
  const float* W1 = task ? W1m : W1r;
  const float* b1 = task ? b1m : b1r;
  const float* W2 = task ? W2m : W2r;
  const float* b2 = task ? b2m : b2r;
  int j = lane;                 // hidden unit
  float hid = b1[j];
  for (int c = 0; c < kOUT; ++c)
    hid += pmean[(size_t)b * kOUT + c] * W1[(size_t)c * 32 + j];
  for (int c = 0; c < kOUT; ++c)
    hid += pmax[(size_t)b * kOUT + c] * W1[(size_t)(kOUT + c) * 32 + j];
  hid = hid > 0.f ? hid : 0.f;
  float r = hid * W2[j];
#pragma unroll
  for (int off = 16; off; off >>= 1) r += __shfl_xor(r, off, 32);
  if (lane == 0) outp[task * kB + b] = r + b2[0];
}

// ---------------------------------------------------------------------------
// Workspace layout (bytes, 256-aligned).  fp32 `out` aliases the dead x0/x1
// staging region (safe: stream-serialized, out init happens after GEMM2).
// ---------------------------------------------------------------------------
constexpr size_t al256(size_t x) { return (x + 255) & ~(size_t)255; }
constexpr size_t OFF_WTF  = 0;
constexpr size_t OFF_WTG  = OFF_WTF  + al256((size_t)kEMB * kEMB * 2);
constexpr size_t OFF_ASRC = OFF_WTG  + al256((size_t)kOUT * kEMB * 2);
constexpr size_t OFF_ADST = OFF_ASRC + al256((size_t)kN * kH * 4);
constexpr size_t OFF_AMAX = OFF_ADST + al256((size_t)kN * kH * 4);
constexpr size_t OFF_DEN  = OFF_AMAX + al256((size_t)kN * kH * 4);
constexpr size_t OFF_PSUM = OFF_DEN  + al256((size_t)kN * kH * 4);
constexpr size_t OFF_PMAX = OFF_PSUM + al256((size_t)kB * kOUT * 4);
constexpr size_t OFF_HBF  = OFF_PMAX + al256((size_t)kB * kOUT * 4);
constexpr size_t OFF_X    = OFF_HBF  + al256((size_t)kN * kOUT * 2);
constexpr size_t OFF_X1   = OFF_X    + al256((size_t)kN * kEMB * 2);
constexpr size_t OFF_OUT  = OFF_X;    // aliases x0/x1 after they are dead

extern "C" void kernel_launch(void* const* d_in, const int* in_sizes, int n_in,
                              void* d_out, int out_size, void* d_ws, size_t ws_size,
                              hipStream_t stream) {
  const int*   code     = (const int*)  d_in[0];
  const int*   ei       = (const int*)  d_in[1];   // [2,E] row-major
  const int*   batch    = (const int*)  d_in[2];
  const float* emb      = (const float*)d_in[3];
  const float* W_feat   = (const float*)d_in[4];
  const float* b_feat   = (const float*)d_in[5];
  const float* W_gat    = (const float*)d_in[6];
  const float* att_src  = (const float*)d_in[7];
  const float* att_dst  = (const float*)d_in[8];
  const float* bias_gat = (const float*)d_in[9];
  const float* W1r = (const float*)d_in[10]; const float* b1r = (const float*)d_in[11];
  const float* W2r = (const float*)d_in[12]; const float* b2r = (const float*)d_in[13];
  const float* W1m = (const float*)d_in[14]; const float* b1m = (const float*)d_in[15];
  const float* W2m = (const float*)d_in[16]; const float* b2m = (const float*)d_in[17];

  char* ws = (char*)d_ws;
  bf16*  WtF   = (bf16*) (ws + OFF_WTF);
  bf16*  WtG   = (bf16*) (ws + OFF_WTG);
  float* a_src = (float*)(ws + OFF_ASRC);
  float* a_dst = (float*)(ws + OFF_ADST);
  float* amax  = (float*)(ws + OFF_AMAX);
  float* denom = (float*)(ws + OFF_DEN);
  float* pmean = (float*)(ws + OFF_PSUM);
  float* pmax  = (float*)(ws + OFF_PMAX);
  bf16*  hbf   = (bf16*) (ws + OFF_HBF);
  bf16*  x0    = (bf16*) (ws + OFF_X);
  bf16*  x1    = (bf16*) (ws + OFF_X1);
  float* outN  = (float*)(ws + OFF_OUT);
  float* res   = (float*)d_out;

  // K0: weight transpose + bf16 convert
  transpose_cvt<<<(kEMB * kEMB + 255) / 256, 256, 0, stream>>>(W_feat, WtF, kEMB, kEMB);
  transpose_cvt<<<(kEMB * kOUT + 255) / 256, 256, 0, stream>>>(W_gat, WtG, kEMB, kOUT);

  // K1: embedding gather -> bf16
  gather_embed<<<(kN * kEMB / 8) / 256, 256, 0, stream>>>(code, emb, x0);

  // K2: x1 = bf16(x0 @ W_feat + b_feat)   [32768,512]x[512,512]
  wmma_gemm_bf16<<<dim3(kN / 32, kEMB / 256), 256, 0, stream>>>(
      x0, WtF, b_feat, x1, kN, kEMB, kEMB);

  // K3: h = bf16(x1 @ W_gat)              [32768,512]x[512,2048]
  wmma_gemm_bf16<<<dim3(kN / 32, kOUT / 256), 256, 0, stream>>>(
      x1, WtG, nullptr, hbf, kN, kOUT, kEMB);

  // K4: attention coefficients (wave per node-head)
  attn_coef<<<(kN * kH) / 8, 256, 0, stream>>>(hbf, att_src, att_dst, a_src, a_dst);

  // K5: init accumulators (out aliases x0/x1, which are now dead)
  init_ws<<<((size_t)kN * kOUT + 255) / 256, 256, 0, stream>>>(outN, amax, denom);

  // K6/K7: softmax over incoming edges (incl. self loops)
  const int ET = (kE + kN) * kH;
  edge_max   <<<(ET + 255) / 256, 256, 0, stream>>>(ei, a_src, a_dst, amax);
  edge_expsum<<<(ET + 255) / 256, 256, 0, stream>>>(ei, a_src, a_dst, amax, denom);

  // K8: weighted aggregation (wave per edge-head; bf16 h gather L2-resident)
  edge_aggregate<<<ET / 8, 256, 0, stream>>>(ei, a_src, a_dst, amax, denom, hbf, outN);

  // K9: atomic-free pooling (+bias, leaky_relu(0.01), mean/max)
  pool_by_graph<<<dim3(kB, kOUT / 256), 256, 0, stream>>>(
      outN, bias_gat, batch, pmean, pmax);

  // K10: readmit / mortality heads
  mlp_heads<<<(kB * 2) / 8, 256, 0, stream>>>(pmean, pmax,
      W1r, b1r, W2r, b2r, W1m, b1m, W2m, b2m, res);
}
